// MaskAttentionSampler_81956565942650
// MI455X (gfx1250) — matrix operand order repaired and checked by reading, not compiled, not run on hardware
//
#include <hip/hip_runtime.h>
#include <hip/hip_bf16.h>
#include <stdint.h>

typedef __attribute__((ext_vector_type(16))) __bf16 v16bf;
typedef __attribute__((ext_vector_type(8)))  float  v8f;
typedef __attribute__((ext_vector_type(4)))  unsigned int v4u;
typedef __attribute__((ext_vector_type(8)))  int    v8i;
typedef __attribute__((ext_vector_type(4)))  int    v4i;

#define NROWS 131072
#define KDIM  512            // codebook size
#define DDIM  512            // feature dim
#define RPB   32             // rows per block (2 row-groups x 16)
#define CHUNK 32             // D-chunk per WMMA pass (K of bf16 wmma)
#define NCHUNK (DDIM / CHUNK)

// LDS layout (bytes, 139264 total):
//   [0      ..  32767]  E-hi chunk buf0 (512 codes x 32 d, bf16)  } aliased by
//   [32768  ..  65535]  E-lo chunk buf0                            } score tile
//   [65536  ..  98303]  E-hi chunk buf1                            } Ss (64 KB)
//   [98304  .. 131071]  E-lo chunk buf1
//   [131072 .. 135167]  X chunk buf0 (32 rows x 32 d, fp32)
//   [135168 .. 139263]  X chunk buf1
#define EH0 0
#define EL0 32768
#define EH1 65536
#define EL1 98304
#define XS0 131072
#define XS1 135168
#define SMEM_BYTES 139264

// ---- prep: one-time split E fp32 -> (hi, lo) bf16 planes in d_ws ----
__global__ __launch_bounds__(256)
void prep_E(const float* __restrict__ E, __bf16* __restrict__ Ehi,
            __bf16* __restrict__ Elo) {
    int i = blockIdx.x * 256 + threadIdx.x;          // over 512*512
    float x = E[i];
    __bf16 h = (__bf16)x;
    Ehi[i] = h;
    Elo[i] = (__bf16)(x - (float)h);
}

// TDM load of Eplane[:, chunk*32 : +32] (bf16, row stride DDIM elements)
// into LDS at lds_off, laid out [code][32 bf16]. D# per ISA 08 §8.3/8.4.
__device__ __forceinline__ void tdm_load_plane(const __bf16* Pg, int chunk,
                                               unsigned lds_off) {
    unsigned long long ga = (unsigned long long)(uintptr_t)Pg
                          + (unsigned long long)chunk * (CHUNK * 2);
    v4u g0;
    g0[0] = 1u;                                      // count=1, no gather
    g0[1] = lds_off;                                 // lds_addr (bytes)
    g0[2] = (unsigned)ga;                            // global_addr[31:0]
    g0[3] = (unsigned)((ga >> 32) & 0x1FFFFFFull)    // global_addr[56:32]
          | (2u << 30);                              // type = 2 (image)
    v8i g1;
    g1[0] = (int)(1u << 16);                         // data_size=2B; mask=0
    g1[1] = (int)((unsigned)(DDIM & 0xFFFF) << 16);  // tensor_dim0[15:0]
    g1[2] = (int)(((DDIM >> 16) & 0xFFFF) | ((KDIM & 0xFFFF) << 16)); // dim0 hi | dim1 lo
    g1[3] = (int)(((KDIM >> 16) & 0xFFFF) | ((unsigned)CHUNK << 16)); // dim1 hi | tile_dim0
    g1[4] = (int)(KDIM & 0xFFFF);                    // tile_dim1=512, tile_dim2=0
    g1[5] = DDIM;                                    // tensor_dim0_stride[31:0]
    g1[6] = 0;                                       // stride0 hi | stride1 lo
    g1[7] = 0;                                       // stride1 hi
    v4i z4 = {0, 0, 0, 0};                           // groups 2/3 unused (2-D)
    v8i z8 = {0, 0, 0, 0, 0, 0, 0, 0};
    __builtin_amdgcn_tensor_load_to_lds(g0, g1, z4, z4, z8, 0);
}

__global__ __launch_bounds__(256)
void masksampler_fused(const float* __restrict__ X,
                       const float* __restrict__ E,
                       const float* __restrict__ G,
                       const __bf16* __restrict__ Ehi,
                       const __bf16* __restrict__ Elo,
                       float* __restrict__ outRec,    // [N, D]
                       float* __restrict__ outDist)   // [N, K]
{
    extern __shared__ char smem[];
    const int tid  = threadIdx.x;
    const int lane = tid & 31;
    const int wv   = tid >> 5;          // 0..7
    const int cg   = wv & 3;            // column group: 128 codes each
    const int rg   = wv >> 2;           // row group: 16 rows each
    const long row0 = (long)blockIdx.x * RPB;
    const unsigned ldsbase = (unsigned)(uintptr_t)smem;  // LDS aperture: low 32b

    // ---- prologue: stage chunk 0 ----
    {   // X chunk -> LDS (coalesced float4, all 256 threads)
        int r  = tid >> 3;
        int dq = (tid & 7) * 4;
        *(float4*)(smem + XS0 + (r * CHUNK + dq) * 4) =
            *(const float4*)(X + (row0 + r) * DDIM + 0 * CHUNK + dq);
    }
    if (wv == 0) {
        tdm_load_plane(Ehi, 0, ldsbase + EH0);
        tdm_load_plane(Elo, 0, ldsbase + EL0);
    }

    v8f acc[8];
    const v8f vz = {0.f, 0.f, 0.f, 0.f, 0.f, 0.f, 0.f, 0.f};
#pragma unroll
    for (int t = 0; t < 8; ++t) acc[t] = vz;

    for (int c = 0; c < NCHUNK; ++c) {
        __builtin_amdgcn_s_wait_tensorcnt(0);
        __syncthreads();                 // chunk c buffers ready; chunk c-1 consumed

        if (c + 1 < NCHUNK) {            // prefetch chunk c+1 into the other buffer
            int r  = tid >> 3;
            int dq = (tid & 7) * 4;
            char* xn = smem + (((c + 1) & 1) ? XS1 : XS0);
            *(float4*)(xn + (r * CHUNK + dq) * 4) =
                *(const float4*)(X + (row0 + r) * DDIM + (c + 1) * CHUNK + dq);
            if (wv == 0) {
                tdm_load_plane(Ehi, c + 1, ldsbase + (((c + 1) & 1) ? EH1 : EH0));
                tdm_load_plane(Elo, c + 1, ldsbase + (((c + 1) & 1) ? EL1 : EL0));
            }
        }

        const float*  xb = (const float*)(smem + ((c & 1) ? XS1 : XS0));
        const __bf16* eh = (const __bf16*)(smem + ((c & 1) ? EH1 : EH0));
        const __bf16* el = (const __bf16*)(smem + ((c & 1) ? EL1 : EL0));

        // A operand 16x32 bf16 hi/lo (lane<16: K 0-15; lane>=16: K 16-31)
        const float* xrow = xb + (rg * 16 + (lane & 15)) * CHUNK;
        const int kb = (lane >> 4) * 16;
        v16bf ahi, alo;
#pragma unroll
        for (int e = 0; e < 8; ++e) {
            float x0 = xrow[kb + e];
            float x1 = xrow[8 + kb + e];
            __bf16 h0 = (__bf16)x0, h1 = (__bf16)x1;
            ahi[e] = h0;  ahi[8 + e] = h1;
            alo[e]     = (__bf16)(x0 - (float)h0);
            alo[8 + e] = (__bf16)(x1 - (float)h1);
        }

        // B operands: raw 32-byte LDS vector loads, no conversion VALU.
        // lane N=lane&15, K = 16*(lane>>4) + e  (contiguous bf16 run)
        const int boff = (lane >> 4) * 16;
#pragma unroll
        for (int t = 0; t < 8; ++t) {
            int code = cg * 128 + t * 16 + (lane & 15);
            v16bf bhi = *(const v16bf*)(eh + code * CHUNK + boff);
            v16bf blo = *(const v16bf*)(el + code * CHUNK + boff);
            // fp32-accurate bf16x2: hi*hi + hi*lo + lo*hi
            acc[t] = __builtin_amdgcn_wmma_f32_16x16x32_bf16(false, ahi, false, bhi, (short)0, acc[t], false, false);
            acc[t] = __builtin_amdgcn_wmma_f32_16x16x32_bf16(false, ahi, false, blo, (short)0, acc[t], false, false);
            acc[t] = __builtin_amdgcn_wmma_f32_16x16x32_bf16(false, alo, false, bhi, (short)0, acc[t], false, false);
        }
    }

    // ---- park scaled scores in LDS (Ss aliases buf0 region, 32x512 fp32) ----
    const float inv_sqrt_d = 0.044194173824159216f;  // 1/sqrt(512)
    float* Ss = (float*)smem;                        // [RPB][KDIM]
    const int mrow = (lane >> 4) * 8;                // C/D: M = vgpr + 8*(lane>=16)
#pragma unroll
    for (int t = 0; t < 8; ++t) {
        int col = cg * 128 + t * 16 + (lane & 15);
#pragma unroll
        for (int i = 0; i < 8; ++i) {
            int r = rg * 16 + mrow + i;
            Ss[r * KDIM + col] = acc[t][i] * inv_sqrt_d;
        }
    }
    __syncthreads();

    // ---- softmax(dist) + gumbel argmax + embedding gather, 4 rows per wave ----
    for (int rr = 0; rr < 4; ++rr) {
        int r = wv * 4 + rr;
        long grow = row0 + r;
        const float* srow = Ss + r * KDIM;

        float v[16];
#pragma unroll
        for (int j = 0; j < 16; ++j) v[j] = srow[lane + 32 * j];

        float mx = v[0];
#pragma unroll
        for (int j = 1; j < 16; ++j) mx = fmaxf(mx, v[j]);
#pragma unroll
        for (int off = 16; off > 0; off >>= 1) mx = fmaxf(mx, __shfl_xor(mx, off, 32));

        float ex[16]; float sum = 0.f;
#pragma unroll
        for (int j = 0; j < 16; ++j) { ex[j] = __expf(v[j] - mx); sum += ex[j]; }
#pragma unroll
        for (int off = 16; off > 0; off >>= 1) sum += __shfl_xor(sum, off, 32);
        float rs = 1.0f / sum;
#pragma unroll
        for (int j = 0; j < 16; ++j)
            outDist[grow * KDIM + lane + 32 * j] = ex[j] * rs;

        // argmax(score + gumbel); tau>0 monotone so /tau is irrelevant
        const float* gp = G + grow * KDIM;
        float bv = -3.0e38f; int bi = 0;
#pragma unroll
        for (int j = 0; j < 16; ++j) {
            int   idx = lane + 32 * j;
            float t   = v[j] + gp[idx];
            if (t > bv) { bv = t; bi = idx; }
        }
#pragma unroll
        for (int off = 16; off > 0; off >>= 1) {
            float ov = __shfl_xor(bv, off, 32);
            int   oi = __shfl_xor(bi, off, 32);
            if (ov > bv || (ov == bv && oi < bi)) { bv = ov; bi = oi; }
        }

        // straight-through forward value == one_hot @ E == exact fp32 row copy
        const float4* er = (const float4*)(E + (long)bi * DDIM);
        float4* orow = (float4*)(outRec + grow * DDIM);
#pragma unroll
        for (int j = 0; j < 4; ++j) orow[lane + 32 * j] = er[lane + 32 * j];
    }
}

extern "C" void kernel_launch(void* const* d_in, const int* in_sizes, int n_in,
                              void* d_out, int out_size, void* d_ws, size_t ws_size,
                              hipStream_t stream) {
    (void)in_sizes; (void)n_in; (void)ws_size; (void)out_size;
    const float* X = (const float*)d_in[0];   // [N, D]
    const float* E = (const float*)d_in[1];   // [K, D]
    const float* G = (const float*)d_in[2];   // [N, K]
    float* outRec  = (float*)d_out;                                   // [N, D]
    float* outDist = (float*)d_out + (size_t)NROWS * DDIM;            // [N, K]

    __bf16* Ehi = (__bf16*)d_ws;                       // 1 MB total in d_ws (2 x 512 KB planes)
    __bf16* Elo = Ehi + (size_t)KDIM * DDIM;

    prep_E<<<dim3((KDIM * DDIM) / 256), dim3(256), 0, stream>>>(E, Ehi, Elo);

    masksampler_fused<<<dim3(NROWS / RPB), dim3(256), SMEM_BYTES, stream>>>(
        X, E, G, Ehi, Elo, outRec, outDist);
}
